// Diffusion_45243185496096
// MI455X (gfx1250) — matrix operand order repaired and checked by reading, not compiled, not run on hardware
//
#include <hip/hip_runtime.h>
#include <math.h>

#define BB 8
#define NN 512
#define FF 32
#define LL 9
#define TI 16

typedef __attribute__((ext_vector_type(16))) _Float16 v16h;
typedef __attribute__((ext_vector_type(8)))  float    v8f;

struct LayerParams {
  const float *msg1W, *msg1b, *msg2W, *msg2b, *fwW, *fwb,
              *up1W, *up1b, *up2W, *up2b,
              *cw1W, *cw1b, *cw2W, *cw2b, *cw3W, *cw3b;
};

// fast hardware transcendentals: v_exp_f32 / v_rcp_f32 / v_sqrt_f32
__device__ __forceinline__ float frcp(float x)  { return __builtin_amdgcn_rcpf(x); }
__device__ __forceinline__ float fsqrt(float x) { return __builtin_amdgcn_sqrtf(x); }
__device__ __forceinline__ float sigmf(float x) { return frcp(1.0f + __expf(-x)); }
__device__ __forceinline__ float siluf(float x) { return x * sigmf(x); }

// sum across the 16 lanes of a half-wave (lanes differing in bits 0..3)
__device__ __forceinline__ float red16(float v) {
  v += __shfl_xor(v, 8, 32);
  v += __shfl_xor(v, 4, 32);
  v += __shfl_xor(v, 2, 32);
  v += __shfl_xor(v, 1, 32);
  return v;
}

// -------- per-node projections: him = H@W1m[0:32], hjm = H@W1m[32:64], same for cw1 --------
__global__ __launch_bounds__(128)
void egnn_node_proj_kernel(const float* __restrict__ Hin, const float* __restrict__ msg1W,
                           const float* __restrict__ cw1W,
                           float* __restrict__ him, float* __restrict__ hjm,
                           float* __restrict__ hic, float* __restrict__ hjc)
{
  __shared__ float hS[FF];
  const int n = blockIdx.x;
  if (threadIdx.x < FF) hS[threadIdx.x] = Hin[n * FF + threadIdx.x];
  __syncthreads();
  const int fo = threadIdx.x & 31;
  const int part = threadIdx.x >> 5;                 // 0:him 1:hjm 2:hic 3:hjc
  const float* W = ((part < 2) ? msg1W : cw1W) + ((part & 1) ? FF * FF : 0);
  float s = 0.f;
#pragma unroll
  for (int k = 0; k < FF; ++k) s += hS[k] * W[k * FF + fo];
  float* dst = (part == 0) ? him : (part == 1) ? hjm : (part == 2) ? hic : hjc;
  dst[n * FF + fo] = s;
}

// -------- fused pairwise layer --------
__global__ __launch_bounds__(256)
void egnn_layer_kernel(const float* __restrict__ Xin, const float* __restrict__ Hin,
                       const float* __restrict__ X0g, const float* __restrict__ Kg,
                       const float* __restrict__ him, const float* __restrict__ hjm,
                       const float* __restrict__ hic, const float* __restrict__ hjc,
                       LayerParams p, float* __restrict__ Xout, float* __restrict__ Hout)
{
  __shared__ float himS[TI * FF], hicS[TI * FF], HiS[TI * FF], u1S[TI * FF], aggS[TI * FF];
  __shared__ float XiS[TI * 3], X0iS[TI * 3], xaccS[TI * 3];
  __shared__ float wmdS[FF], wm0S[FF], bm1S[FF], wcdS[FF], wc0S[FF], bc1S[FF];
  __shared__ float fwS[FF], cw3S[FF], bm2S[FF], bc2S[FF], up1bS[FF], up2bS[FF];
  __shared__ float sclr[2];
  __shared__ float up1S[2 * FF * FF], up2S[FF * FF];
  __shared__ float hjmS[8][TI * FF], hjcS[8][TI * FF];
  __shared__ float XjS[8][TI * 3], X0jS[8][TI * 3];

  const int tid  = threadIdx.x;
  const int lane = tid & 31, wid = tid >> 5;
  const int b  = blockIdx.x >> 5;               // NN/TI == 32 i-tiles per batch
  const int i0 = (blockIdx.x & 31) * TI;
  const int nodeBase = b * NN + i0;

  for (int q = tid; q < TI * FF; q += 256) {
    int r = q >> 5, c = q & 31;
    himS[q] = him[(nodeBase + r) * FF + c];
    hicS[q] = hic[(nodeBase + r) * FF + c];
    HiS[q]  = Hin[(nodeBase + r) * FF + c];
    aggS[q] = 0.f;
  }
  for (int q = tid; q < TI * 3; q += 256) {
    int r = q / 3, c = q % 3;
    XiS[q]   = Xin[(nodeBase + r) * 3 + c];
    X0iS[q]  = X0g[(nodeBase + r) * 3 + c];
    xaccS[q] = 0.f;
  }
  if (tid < FF) {
    wmdS[tid] = p.msg1W[(2 * FF) * FF + tid];      // dsq row
    wm0S[tid] = p.msg1W[(2 * FF + 1) * FF + tid];  // d0sq row
    bm1S[tid] = p.msg1b[tid];
    wcdS[tid] = p.cw1W[(2 * FF) * FF + tid];
    wc0S[tid] = p.cw1W[(2 * FF + 1) * FF + tid];
    bc1S[tid] = p.cw1b[tid];
    fwS[tid]  = p.fwW[tid];
    cw3S[tid] = p.cw3W[tid];
    bm2S[tid] = p.msg2b[tid];
    bc2S[tid] = p.cw2b[tid];
    up1bS[tid] = p.up1b[tid];
    up2bS[tid] = p.up2b[tid];
  }
  if (tid == 0) { sclr[0] = p.fwb[0]; sclr[1] = p.cw3b[0]; }
  for (int q = tid; q < 2 * FF * FF; q += 256) up1S[q] = p.up1W[q];
  for (int q = tid; q < FF * FF; q += 256)     up2S[q] = p.up2W[q];

  // wave-resident f16 B fragments (32x16, K = t + (t&8) + 8*hi, col = lane&15)
  const int hi = lane >> 4, fcol = lane & 15;
  v16h bM0, bM1, bC0, bC1;
#pragma unroll
  for (int t = 0; t < 16; ++t) {
    int k = t + (t & 8) + (hi << 3);
    bM0[t] = (_Float16)p.msg2W[k * FF + fcol];
    bM1[t] = (_Float16)p.msg2W[k * FF + 16 + fcol];
    bC0[t] = (_Float16)p.cw2W[k * FF + fcol];
    bC1[t] = (_Float16)p.cw2W[k * FF + 16 + fcol];
  }
  __syncthreads();
  const float fwb = sclr[0], cb3 = sclr[1];
  float* hjmW = hjmS[wid]; float* hjcW = hjcS[wid];
  float* XjW  = XjS[wid];  float* X0jW = X0jS[wid];

  for (int jt = wid; jt < NN / TI; jt += 8) {
    const int j0 = jt * TI;
    for (int q = lane; q < TI * FF; q += 32) {
      int r = q >> 5, c = q & 31;
      hjmW[q] = hjm[(b * NN + j0 + r) * FF + c];
      hjcW[q] = hjc[(b * NN + j0 + r) * FF + c];
    }
    for (int q = lane; q < TI * 3; q += 32) {
      int r = q / 3, c = q % 3;
      XjW[q]  = Xin[(b * NN + j0 + r) * 3 + c];
      X0jW[q] = X0g[(b * NN + j0 + r) * 3 + c];
    }
    // same-wave LDS staging: make stores visible to all lanes of this wave
    __builtin_amdgcn_wave_barrier();
    asm volatile("s_wait_dscnt 0" ::: "memory");
    __builtin_amdgcn_wave_barrier();

    for (int m = 0; m < TI; ++m) {
      const int jj = fcol;
      float dx = XiS[m * 3 + 0] - XjW[jj * 3 + 0];
      float dy = XiS[m * 3 + 1] - XjW[jj * 3 + 1];
      float dz = XiS[m * 3 + 2] - XjW[jj * 3 + 2];
      float dsq = dx * dx + dy * dy + dz * dz;
      float ex = X0iS[m * 3 + 0] - X0jW[jj * 3 + 0];
      float ey = X0iS[m * 3 + 1] - X0jW[jj * 3 + 1];
      float ez = X0iS[m * 3 + 2] - X0jW[jj * 3 + 2];
      float d0q = ex * ex + ey * ey + ez * ez;

      // ---------- message branch: MH = silu(silu(pre) @ W2 + b2) ----------
      v16h aM;
#pragma unroll
      for (int t = 0; t < 16; ++t) {
        int k = t + (t & 8) + (hi << 3);          // 16-bit A 16x32 lane layout
        float pre = himS[m * FF + k] + hjmW[jj * FF + k] +
                    dsq * wmdS[k] + d0q * wm0S[k] + bm1S[k];
        aM[t] = (_Float16)siluf(pre);
      }
      v8f c0, c1;
#pragma unroll
      for (int r = 0; r < 8; ++r) { c0[r] = bm2S[fcol]; c1[r] = bm2S[16 + fcol]; }
      c0 = __builtin_amdgcn_wmma_f32_16x16x32_f16(false, aM, false, bM0, (short)0, c0, false, false);
      c1 = __builtin_amdgcn_wmma_f32_16x16x32_f16(false, aM, false, bM1, (short)0, c1, false, false);

      float mh0[8], mh1[8], kv[8], wh[8];
#pragma unroll
      for (int r = 0; r < 8; ++r) { mh0[r] = siluf(c0[r]); mh1[r] = siluf(c1[r]); }
#pragma unroll
      for (int r = 0; r < 8; ++r) {
        int rowjj = r + (hi << 3);                // D layout: M = r + 8*hi, N = lane&15
        kv[r] = Kg[(size_t)(b * NN + i0 + m) * NN + (j0 + rowjj)];
        float s = red16(mh0[r] * fwS[fcol] + mh1[r] * fwS[16 + fcol]);
        wh[r] = sigmf(s + fwb) * kv[r];
      }
      float a0 = 0.f, a1 = 0.f;
#pragma unroll
      for (int r = 0; r < 8; ++r) { a0 += mh0[r] * wh[r]; a1 += mh1[r] * wh[r]; }
      a0 += __shfl_xor(a0, 16, 32);
      a1 += __shfl_xor(a1, 16, 32);
      if (lane < 16) {
        atomicAdd(&aggS[m * FF + lane], a0);
        atomicAdd(&aggS[m * FF + 16 + lane], a1);
      }

      // ---------- coordinate branch: WX = (silu(silu(cpre)@Wc2+bc2)@cw3+cb3)*K ----------
      v16h aC;
#pragma unroll
      for (int t = 0; t < 16; ++t) {
        int k = t + (t & 8) + (hi << 3);
        float pre = hicS[m * FF + k] + hjcW[jj * FF + k] +
                    dsq * wcdS[k] + d0q * wc0S[k] + bc1S[k];
        aC[t] = (_Float16)siluf(pre);
      }
      v8f e0, e1;
#pragma unroll
      for (int r = 0; r < 8; ++r) { e0[r] = bc2S[fcol]; e1[r] = bc2S[16 + fcol]; }
      e0 = __builtin_amdgcn_wmma_f32_16x16x32_f16(false, aC, false, bC0, (short)0, e0, false, false);
      e1 = __builtin_amdgcn_wmma_f32_16x16x32_f16(false, aC, false, bC1, (short)0, e1, false, false);

      float wx[8];
#pragma unroll
      for (int r = 0; r < 8; ++r) {
        float s0 = siluf(e0[r]), s1 = siluf(e1[r]);
        float s = red16(s0 * cw3S[fcol] + s1 * cw3S[16 + fcol]);
        wx[r] = (s + cb3) * kv[r];
      }
      if (fcol == 0) {                            // lanes 0 and 16 cover rows 0-7 / 8-15
        float ax = 0.f, ay = 0.f, az = 0.f;
#pragma unroll
        for (int r = 0; r < 8; ++r) {
          int rowjj = r + (hi << 3);
          float rx = XiS[m * 3 + 0] - XjW[rowjj * 3 + 0];
          float ry = XiS[m * 3 + 1] - XjW[rowjj * 3 + 1];
          float rz = XiS[m * 3 + 2] - XjW[rowjj * 3 + 2];
          float dq = rx * rx + ry * ry + rz * rz;
          float w = wx[r] * frcp(fsqrt(dq) + 1.0f);
          ax += rx * w; ay += ry * w; az += rz * w;
        }
        atomicAdd(&xaccS[m * 3 + 0], ax);
        atomicAdd(&xaccS[m * 3 + 1], ay);
        atomicAdd(&xaccS[m * 3 + 2], az);
      }
    }
  }
  __syncthreads();

  // ---------- residual updates ----------
  {
    int node = tid >> 4, f = tid & 15;
#pragma unroll
    for (int g = 0; g < 2; ++g) {
      int fo = f + g * 16;
      float s = up1bS[fo];
      for (int k = 0; k < FF; ++k) {
        s += HiS[node * FF + k]  * up1S[k * FF + fo];
        s += aggS[node * FF + k] * up1S[(FF + k) * FF + fo];
      }
      u1S[node * FF + fo] = siluf(s);
    }
  }
  __syncthreads();
  {
    int node = tid >> 4, f = tid & 15;
#pragma unroll
    for (int g = 0; g < 2; ++g) {
      int fo = f + g * 16;
      float s = up2bS[fo] + HiS[node * FF + fo];
      for (int k = 0; k < FF; ++k) s += u1S[node * FF + k] * up2S[k * FF + fo];
      Hout[(nodeBase + node) * FF + fo] = s;
    }
    if (tid < TI * 3) {
      int nd = tid / 3, c = tid % 3;
      Xout[(nodeBase + nd) * 3 + c] = XiS[nd * 3 + c] + xaccS[nd * 3 + c];
    }
  }
}

__global__ void egnn_concat_kernel(const float* __restrict__ Xf, const float* __restrict__ Hf,
                                   float* __restrict__ out)
{
  int idx = blockIdx.x * blockDim.x + threadIdx.x;
  if (idx >= BB * NN * 35) return;
  int c = idx % 35, n = idx / 35;
  out[idx] = (c < 3) ? Xf[n * 3 + c] : Hf[n * FF + (c - 3)];
}

extern "C" void kernel_launch(void* const* d_in, const int* in_sizes, int n_in,
                              void* d_out, int out_size, void* d_ws, size_t ws_size,
                              hipStream_t stream) {
  (void)in_sizes; (void)n_in; (void)out_size; (void)ws_size;
  const float* X  = (const float*)d_in[0];
  const float* H  = (const float*)d_in[1];
  const float* Kg = (const float*)d_in[2];

  float* ws = (float*)d_ws;
  const size_t nx = (size_t)BB * NN * 3;   // 12288
  const size_t nh = (size_t)BB * NN * FF;  // 131072
  float* X0   = ws;
  float* Xb0  = X0 + nx;
  float* Xb1  = Xb0 + nx;
  float* Hb0  = Xb1 + nx;
  float* Hb1  = Hb0 + nh;
  float* himB = Hb1 + nh;
  float* hjmB = himB + nh;
  float* hicB = hjmB + nh;
  float* hjcB = hicB + nh;
  float* Xb[2] = { Xb0, Xb1 };
  float* Hb[2] = { Hb0, Hb1 };

  hipMemcpyAsync(X0,    X, nx * sizeof(float), hipMemcpyDeviceToDevice, stream);
  hipMemcpyAsync(Xb[0], X, nx * sizeof(float), hipMemcpyDeviceToDevice, stream);
  hipMemcpyAsync(Hb[0], H, nh * sizeof(float), hipMemcpyDeviceToDevice, stream);

  for (int l = 0; l < LL; ++l) {
    // params flattened in setup_inputs() insertion order:
    // msg1(W,b) msg2(W,b) fw(W,b) up1(W,b) up2(W,b) cw1(W,b) cw2(W,b) cw3(W,b)
    const int base = 3 + l * 16;
    LayerParams p;
    p.msg1W = (const float*)d_in[base + 0];  p.msg1b = (const float*)d_in[base + 1];
    p.msg2W = (const float*)d_in[base + 2];  p.msg2b = (const float*)d_in[base + 3];
    p.fwW   = (const float*)d_in[base + 4];  p.fwb   = (const float*)d_in[base + 5];
    p.up1W  = (const float*)d_in[base + 6];  p.up1b  = (const float*)d_in[base + 7];
    p.up2W  = (const float*)d_in[base + 8];  p.up2b  = (const float*)d_in[base + 9];
    p.cw1W  = (const float*)d_in[base + 10]; p.cw1b  = (const float*)d_in[base + 11];
    p.cw2W  = (const float*)d_in[base + 12]; p.cw2b  = (const float*)d_in[base + 13];
    p.cw3W  = (const float*)d_in[base + 14]; p.cw3b  = (const float*)d_in[base + 15];

    egnn_node_proj_kernel<<<BB * NN, 128, 0, stream>>>(
        Hb[l & 1], p.msg1W, p.cw1W, himB, hjmB, hicB, hjcB);
    egnn_layer_kernel<<<BB * (NN / TI), 256, 0, stream>>>(
        Xb[l & 1], Hb[l & 1], X0, Kg, himB, hjmB, hicB, hjcB, p,
        Xb[(l + 1) & 1], Hb[(l + 1) & 1]);
  }
  egnn_concat_kernel<<<(BB * NN * 35 + 255) / 256, 256, 0, stream>>>(
      Xb[LL & 1], Hb[LL & 1], (float*)d_out);
}
